// BiLSTMClassifier_83219286327902
// MI455X (gfx1250) — compile-verified, matrix operand order checked by reading, not run on hardware
//
#include <hip/hip_runtime.h>
#include <hip/hip_bf16.h>

#define BB   16
#define SS   2048
#define EE   128
#define HDIM 512
#define G4   2048            // 4*HD
#define KTOT 640             // E + HD
#define ASTR 648             // padded LDS row stride (shorts): 1296B rows, 16B aligned
#define NWGD 8               // workgroups per direction

typedef __attribute__((ext_vector_type(16))) __bf16 v16bf;
typedef __attribute__((ext_vector_type(8)))  __bf16 v8bf;
typedef __attribute__((ext_vector_type(8)))  float  v8f;

__device__ __forceinline__ unsigned short f2bf(float f) {
    unsigned u; __builtin_memcpy(&u, &f, 4);
    u += 0x7FFFu + ((u >> 16) & 1u);           // round-to-nearest-even
    return (unsigned short)(u >> 16);
}
__device__ __forceinline__ float bf2f(unsigned short s) {
    unsigned u = ((unsigned)s) << 16; float f; __builtin_memcpy(&f, &u, 4); return f;
}
// branch-free activations: v_exp_f32 + v_rcp_f32 only (no libm slow paths)
__device__ __forceinline__ float fsigm(float x) {
    return __builtin_amdgcn_rcpf(1.0f + __expf(-x));
}
__device__ __forceinline__ float ftanh(float x) {
    float e = __expf(2.0f * x);                // inf for large x -> rcp 0 -> +1
    return 1.0f - 2.0f * __builtin_amdgcn_rcpf(e + 1.0f);
}

// ---------------------------------------------------------------- init: zero h state + barrier
__global__ void init_kernel(unsigned short* __restrict__ hbuf, unsigned* __restrict__ bar) {
    int i = blockIdx.x * blockDim.x + threadIdx.x;
    for (int k = i; k < 2 * BB * HDIM; k += blockDim.x * gridDim.x) hbuf[k] = 0;
    if (i < 4) bar[i] = 0u;
}

// ---------------------------------------------------------------- embedding gather -> bf16
__global__ void gather_cast_kernel(const int* __restrict__ ids,
                                   const float* __restrict__ embed,
                                   unsigned short* __restrict__ xb) {
    const size_t N = (size_t)BB * SS * EE;
    for (size_t i = (size_t)blockIdx.x * blockDim.x + threadIdx.x; i < N;
         i += (size_t)gridDim.x * blockDim.x) {
        size_t bs = i / EE; int e = (int)(i - bs * EE);
        int tok = ids[bs];
        xb[i] = f2bf(embed[(size_t)tok * EE + e]);
    }
}

// ---------------------------------------------------------------- fuse+transpose weights:
// wt[d][n][k] = (k<E ? Wx_d[k][n] : Wh_d[k-E][n]) as bf16  (N-major => contiguous K per lane)
__global__ void wprep_kernel(const float* __restrict__ Wx_f, const float* __restrict__ Wh_f,
                             const float* __restrict__ Wx_b, const float* __restrict__ Wh_b,
                             unsigned short* __restrict__ wt) {
    const size_t N = (size_t)2 * G4 * KTOT;
    for (size_t i = (size_t)blockIdx.x * blockDim.x + threadIdx.x; i < N;
         i += (size_t)gridDim.x * blockDim.x) {
        int d   = (int)(i / ((size_t)G4 * KTOT));
        int rem = (int)(i - (size_t)d * G4 * KTOT);
        int n = rem / KTOT, k = rem - n * KTOT;
        const float* Wx = d ? Wx_b : Wx_f;
        const float* Wh = d ? Wh_b : Wh_f;
        float v = (k < EE) ? Wx[(size_t)k * G4 + n] : Wh[(size_t)(k - EE) * G4 + n];
        wt[i] = f2bf(v);
    }
}

// ---------------------------------------------------------------- recurrent scan (persistent)
// grid = 16 blocks (8 per direction) x 128 threads (4 waves); wave w owns hd cols
// [j*64 + w*16, +16) and computes its i/f/g/o 16x16 tiles with bf16 WMMA, K=640.
__global__ void __launch_bounds__(128) bilstm_scan_kernel(
        const unsigned short* __restrict__ xb,     // [B][S][E] bf16
        const unsigned short* __restrict__ wt,     // [2][2048][640] bf16
        const int* __restrict__ amask,             // [B][S]
        const float* __restrict__ bf_, const float* __restrict__ bb_,
        unsigned short* __restrict__ hbuf,         // [2][B][HD] bf16 (carry h)
        unsigned short* __restrict__ outcat,       // [B][S][2*HD] bf16 (raw outputs)
        float* __restrict__ hfin,                  // [2][B][HD] f32 (final carry)
        unsigned* __restrict__ bar)                // [2] x {cnt, gen}
{
    __shared__ __align__(16) unsigned short a_sh[BB * ASTR];
    __shared__ int msk_sh[BB];

    const int tid  = threadIdx.x;
    const int d    = blockIdx.x >> 3;
    const int j    = blockIdx.x & 7;
    const int wave = tid >> 5, lane = tid & 31;
    const int row   = lane & 15;
    const int khalf = (lane < 16) ? 0 : 8;
    const int hd0   = j * 64 + wave * 16;
    const int hd    = hd0 + row;                // this lane's output column
    const float* bias = d ? bb_ : bf_;
    const unsigned short* wtd = wt + (size_t)d * G4 * KTOT;
    unsigned short* hb = hbuf + d * BB * HDIM;
    volatile unsigned* cnt = bar + d * 2;
    volatile unsigned* gen = bar + d * 2 + 1;

    // loop-invariant per-lane bias values (i/f/g/o)
    const float bi = bias[hd];
    const float bfv = bias[HDIM + hd];
    const float bg = bias[2 * HDIM + hd];
    const float bo = bias[3 * HDIM + hd];

    float creg[8];
#pragma unroll
    for (int r = 0; r < 8; ++r) creg[r] = 0.f;

    for (int t = 0; t < SS; ++t) {
        const int tsrc = d ? (SS - 1 - t) : t;

        // ---- phase 1: stage A = [x_t | h] into LDS with b128 copies ----
        // 16 rows x 80 16-byte chunks (x: chunks 0..15, h: chunks 16..79)
        for (int idx = tid; idx < BB * (KTOT / 8); idx += 128) {
            int b  = idx / (KTOT / 8);
            int k  = (idx - b * (KTOT / 8)) * 8;
            const unsigned short* src = (k < EE)
                ? &xb[((size_t)b * SS + tsrc) * EE + k]
                : &hb[b * HDIM + (k - EE)];
            *(uint4*)&a_sh[b * ASTR + k] = *(const uint4*)src;
        }
        if (tid < BB) msk_sh[tid] = amask[tid * SS + tsrc];
        __syncthreads();

        // ---- phase 2: i/f/g/o tiles = A x Wt via v_wmma_f32_16x16x32_bf16 ----
        v8f acc[4];
#pragma unroll
        for (int g = 0; g < 4; ++g)
#pragma unroll
            for (int i = 0; i < 8; ++i) acc[g][i] = 0.f;

        for (int kb = 0; kb < KTOT; kb += 32) {
            const v8bf alo = *(const v8bf*)&a_sh[row * ASTR + kb + khalf];
            const v8bf ahi = *(const v8bf*)&a_sh[row * ASTR + kb + 16 + khalf];
            v16bf afrag;
#pragma unroll
            for (int i = 0; i < 8; ++i) { afrag[i] = alo[i]; afrag[8 + i] = ahi[i]; }
#pragma unroll
            for (int g = 0; g < 4; ++g) {
                const int ncol = g * HDIM + hd0 + row;
                const v8bf blo = *(const v8bf*)&wtd[(size_t)ncol * KTOT + kb + khalf];
                const v8bf bhi = *(const v8bf*)&wtd[(size_t)ncol * KTOT + kb + 16 + khalf];
                v16bf bfrag;
#pragma unroll
                for (int i = 0; i < 8; ++i) { bfrag[i] = blo[i]; bfrag[8 + i] = bhi[i]; }
                acc[g] = __builtin_amdgcn_wmma_f32_16x16x32_bf16(
                             false, afrag, false, bfrag, (short)0, acc[g], false, false);
            }
        }
        __syncthreads();

        // ---- phase 3: LSTM cell, branch-free, entirely in registers ----
#pragma unroll
        for (int r = 0; r < 8; ++r) {
            const int b = r + ((lane < 16) ? 0 : 8);    // C layout: vgpr r -> row r / r+8
            float iv = acc[0][r] + bi;
            float fv = acc[1][r] + bfv;
            float gv = acc[2][r] + bg;
            float ov = acc[3][r] + bo;
            float cn = fsigm(fv) * creg[r] + fsigm(iv) * ftanh(gv);
            float hn = fsigm(ov) * ftanh(cn);
            const bool m = msk_sh[b] > 0;
            creg[r] = m ? cn : creg[r];
            float hold = bf2f(a_sh[b * ASTR + EE + hd]);
            float hk   = m ? hn : hold;                 // masked carry
            hb[b * HDIM + hd] = f2bf(hk);
            const int col = d ? (HDIM + hd) : hd;       // bwd outputs un-flip to tsrc
            outcat[((size_t)b * SS + tsrc) * (2 * HDIM) + col] = f2bf(hn);  // raw output
            if (t == SS - 1) hfin[d * BB * HDIM + b * HDIM + hd] = hk;
        }

        // ---- global sense-reversing barrier across this direction's 8 WGs ----
        __threadfence();
        __syncthreads();
        if (tid == 0) {
            unsigned g0  = *gen;
            unsigned old = atomicAdd((unsigned*)cnt, 1u);
            if (old == NWGD - 1) { *cnt = 0u; __threadfence(); atomicAdd((unsigned*)gen, 1u); }
            else { while (*gen == g0) { __builtin_amdgcn_s_sleep(1); } }
        }
        __syncthreads();
        __builtin_amdgcn_fence(__ATOMIC_ACQUIRE, "agent");  // invalidate stale h in WGP$
    }
}

// ---------------------------------------------------------------- attention logits (+bias)
__global__ void logits_kernel(const float* __restrict__ hfin,
                              const unsigned short* __restrict__ outcat,
                              const int* __restrict__ amask,
                              float* __restrict__ lg, float* __restrict__ dout) {
    int i = blockIdx.x * blockDim.x + threadIdx.x;
    if (i >= BB * SS) return;
    int b = i / SS;
    const unsigned short* o = outcat + (size_t)i * (2 * HDIM);
    float acc = 0.f;
    for (int h = 0; h < HDIM; ++h) acc += hfin[b * HDIM + h] * bf2f(o[h]);
    for (int h = 0; h < HDIM; ++h) acc += hfin[BB * HDIM + b * HDIM + h] * bf2f(o[HDIM + h]);
    acc *= 0.03125f;                                   // 1/sqrt(1024)
    float v = acc + ((amask[i] > 0) ? 0.f : -1e10f);
    lg[i]        = v;
    dout[32 + i] = v;                                  // attn_logits output
}

// ---------------------------------------------------------------- softmax over S per batch
__global__ void softmax_kernel(const float* __restrict__ lg, float* __restrict__ w) {
    __shared__ float red[256];
    int b = blockIdx.x, tid = threadIdx.x;
    const float* x = lg + (size_t)b * SS;
    float mx = -3.4e38f;
    for (int s = tid; s < SS; s += 256) mx = fmaxf(mx, x[s]);
    red[tid] = mx; __syncthreads();
    for (int o = 128; o > 0; o >>= 1) { if (tid < o) red[tid] = fmaxf(red[tid], red[tid + o]); __syncthreads(); }
    mx = red[0]; __syncthreads();
    float sum = 0.f;
    for (int s = tid; s < SS; s += 256) { float e = __expf(x[s] - mx); w[(size_t)b * SS + s] = e; sum += e; }
    red[tid] = sum; __syncthreads();
    for (int o = 128; o > 0; o >>= 1) { if (tid < o) red[tid] += red[tid + o]; __syncthreads(); }
    float inv = 1.f / red[0];
    for (int s = tid; s < SS; s += 256) w[(size_t)b * SS + s] *= inv;
}

// ---------------------------------------------------------------- attn_out = sum_s w * out
__global__ void attnout_kernel(const float* __restrict__ w,
                               const unsigned short* __restrict__ outcat,
                               float* __restrict__ attn) {
    int i = blockIdx.x * blockDim.x + threadIdx.x;
    if (i >= BB * 2 * HDIM) return;
    int b = i / (2 * HDIM), h = i - b * 2 * HDIM;
    float acc = 0.f;
    for (int s = 0; s < SS; ++s)
        acc += w[(size_t)b * SS + s] * bf2f(outcat[((size_t)b * SS + s) * (2 * HDIM) + h]);
    attn[i] = acc;
}

// ---------------------------------------------------------------- classifier head
__global__ void cls_kernel(const float* __restrict__ attn, const float* __restrict__ W_out,
                           const float* __restrict__ b_out, float* __restrict__ dout) {
    int i = threadIdx.x;
    if (i >= BB * 2) return;
    int b = i >> 1, c = i & 1;
    float acc = b_out[c];
    for (int h = 0; h < 2 * HDIM; ++h) acc += attn[b * 2 * HDIM + h] * W_out[h * 2 + c];
    dout[i] = acc;
}

// ================================================================ launch
extern "C" void kernel_launch(void* const* d_in, const int* in_sizes, int n_in,
                              void* d_out, int out_size, void* d_ws, size_t ws_size,
                              hipStream_t stream) {
    const int*   ids   = (const int*)  d_in[0];
    const int*   amask = (const int*)  d_in[1];
    const float* embed = (const float*)d_in[2];
    const float* Wx_f  = (const float*)d_in[3];
    const float* Wh_f  = (const float*)d_in[4];
    const float* b_f   = (const float*)d_in[5];
    const float* Wx_b  = (const float*)d_in[6];
    const float* Wh_b  = (const float*)d_in[7];
    const float* b_b   = (const float*)d_in[8];
    const float* W_out = (const float*)d_in[9];
    const float* b_out = (const float*)d_in[10];
    float* out = (float*)d_out;

    char* ws = (char*)d_ws; size_t off = 0;
    auto carve = [&](size_t bytes) -> void* {
        void* p = ws + off; off = (off + bytes + 255) & ~(size_t)255; return p;
    };
    unsigned short* xb     = (unsigned short*)carve((size_t)BB * SS * EE * 2);        //  8 MB
    unsigned short* wt     = (unsigned short*)carve((size_t)2 * G4 * KTOT * 2);       //  5 MB
    unsigned short* outcat = (unsigned short*)carve((size_t)BB * SS * 2 * HDIM * 2);  // 64 MB
    unsigned short* hbuf   = (unsigned short*)carve((size_t)2 * BB * HDIM * 2);
    float*          hfin   = (float*)         carve((size_t)2 * BB * HDIM * 4);
    float*          lgbuf  = (float*)         carve((size_t)BB * SS * 4);
    float*          wbuf   = (float*)         carve((size_t)BB * SS * 4);
    float*          attn   = (float*)         carve((size_t)BB * 2 * HDIM * 4);
    unsigned*       bar    = (unsigned*)      carve(256);

    init_kernel<<<64, 256, 0, stream>>>(hbuf, bar);
    gather_cast_kernel<<<4096, 256, 0, stream>>>(ids, embed, xb);
    wprep_kernel<<<2560, 256, 0, stream>>>(Wx_f, Wh_f, Wx_b, Wh_b, wt);
    bilstm_scan_kernel<<<16, 128, 0, stream>>>(xb, wt, amask, b_f, b_b,
                                               hbuf, outcat, hfin, bar);
    logits_kernel<<<(BB * SS) / 256, 256, 0, stream>>>(hfin, outcat, amask, lgbuf, out);
    softmax_kernel<<<BB, 256, 0, stream>>>(lgbuf, wbuf);
    attnout_kernel<<<(BB * 2 * HDIM) / 256, 256, 0, stream>>>(wbuf, outcat, attn);
    cls_kernel<<<1, 32, 0, stream>>>(attn, W_out, b_out, out);

    (void)in_sizes; (void)n_in; (void)out_size; (void)ws_size;
}
